// AttentionFusion_13889924235277
// MI455X (gfx1250) — matrix operand order repaired and checked by reading, not compiled, not run on hardware
//
#include <hip/hip_runtime.h>
#include <hip/hip_bf16.h>

#define Bdim 32
#define Ndim 1024
#define CDim 896
#define GDim 384
#define HDim 256
#define ODim 1280

typedef __attribute__((ext_vector_type(16))) __bf16    v16bf_t;
typedef __attribute__((ext_vector_type(8)))  float     v8f_t;
typedef __attribute__((ext_vector_type(8)))  unsigned  v8u_t;
typedef __attribute__((ext_vector_type(4)))  unsigned  v4u_t;
typedef __attribute__((ext_vector_type(8)))  int       v8i_t;
typedef __attribute__((ext_vector_type(4)))  int       v4i_t;

__device__ __forceinline__ unsigned short f2bf(float f) {
    unsigned u = __float_as_uint(f);
    u += 0x7FFFu + ((u >> 16) & 1u);          // round-to-nearest-even
    return (unsigned short)(u >> 16);
}
__device__ __forceinline__ float bf2f(unsigned short h) {
    return __uint_as_float(((unsigned)h) << 16);
}

// ---------------------------------------------------------------------------
// TDM: 2D tile load Global->LDS (bf16 elements). D# packing per CDNA5 ISA §8:
//  group0: [1:0]=count=1 | lds_addr | global_addr[56:0] | type=2 (bits 127:126)
//  group1: data_size=2B; tensor_dim0/1; tile_dim0/1; tensor_dim0_stride
// Issued per-wave (EXEC ignored); tracked with TENSORcnt.
// ---------------------------------------------------------------------------
__device__ __forceinline__ void tdm_load_2d(unsigned lds_off, const void* gptr,
                                            unsigned tile_d0, unsigned tile_d1,
                                            unsigned tensor_d0, unsigned tensor_d1,
                                            unsigned stride0)
{
    unsigned long long ga = (unsigned long long)gptr;
    v4u_t g0;
    g0[0] = 1u;                                          // count=1, user descriptor
    g0[1] = lds_off;                                     // LDS byte address
    g0[2] = (unsigned)(ga & 0xFFFFFFFFull);              // global addr [31:0]
    g0[3] = (unsigned)((ga >> 32) & 0x01FFFFFFull) | (2u << 30); // [56:32] | type=2
    v8i_t g1;
    g1[0] = (int)0x00010000u;                            // data_size=1 (2 bytes)
    g1[1] = (int)((tensor_d0 & 0xFFFFu) << 16);          // tensor_dim0[15:0]
    g1[2] = (int)((tensor_d0 >> 16) | ((tensor_d1 & 0xFFFFu) << 16));
    g1[3] = (int)((tensor_d1 >> 16) | (tile_d0 << 16));  // tile_dim0
    g1[4] = (int)tile_d1;                                // tile_dim1 (tile_dim2=0)
    g1[5] = (int)stride0;                                // tensor_dim0_stride[31:0]
    g1[6] = 0; g1[7] = 0;
    v4i_t z4 = {};
#if defined(__clang_major__) && (__clang_major__ >= 23)
    v8i_t z8 = {};
    __builtin_amdgcn_tensor_load_to_lds(g0, g1, z4, z4, z8, 0);
#else
    __builtin_amdgcn_tensor_load_to_lds(g0, g1, z4, z4, 0);
#endif
}

__device__ __forceinline__ unsigned lds_offset_of(const void* p) {
    // generic pointer to LDS carries the LDS byte offset in its low 32 bits
    return (unsigned)(unsigned long long)p;
}

// ---------------------------------------------------------------------------
// Generic bf16 WMMA GEMM: out[M,Nd] = bf16(A[M,Kd] @ W[Kd,Nd] + bias)
// Block = 64x64 tile, 8 waves, each wave: 16(M) x 32(N) -> 2 accumulators.
// fp32->bf16 conversion happens while staging tiles into LDS (TDM can't
// convert, and a separate conversion pass would add ~250MB of HBM traffic).
// ---------------------------------------------------------------------------
__global__ __launch_bounds__(256) void gemm_bf16_kernel(
    const float* __restrict__ A, const float* __restrict__ W,
    const float* __restrict__ bias, unsigned short* __restrict__ out,
    int M, int Kd, int Nd)
{
    __shared__ unsigned short Alds[64][32];   // [m][k]
    __shared__ unsigned short Blds[64][32];   // [n][k]  (transposed for dword frag reads)
    const int tid  = threadIdx.x;
    const int lane = tid & 31, wave = tid >> 5;
    const int m0 = blockIdx.x * 64, n0 = blockIdx.y * 64;
    const int waveM = wave >> 1, waveN = wave & 1;
    const int g = lane >> 4, hx = lane & 15;

    v8f_t acc0 = {}; v8f_t acc1 = {};

    for (int k0 = 0; k0 < Kd; k0 += 32) {
        __syncthreads();
#pragma unroll
        for (int i = 0; i < 8; ++i) {         // A tile 64x32, fp32->bf16
            int idx = tid + 256 * i;
            int m = idx >> 5, k = idx & 31;
            Alds[m][k] = f2bf(A[(size_t)(m0 + m) * Kd + k0 + k]);
        }
#pragma unroll
        for (int i = 0; i < 8; ++i) {         // W tile 32x64 -> Blds[n][k]
            int idx = tid + 256 * i;
            int k = idx >> 6, n = idx & 63;
            Blds[n][k] = f2bf(W[(size_t)(k0 + k) * Nd + n0 + n]);
        }
        __syncthreads();

        // A fragment: 16x32 bf16 in contiguous k = 16*g + s order (lane halves = K sub-blocks)
        v8u_t a;
#pragma unroll
        for (int v = 0; v < 8; ++v) {
            int kk = 16 * g + 2 * v;
            a[v] = *(const unsigned*)&Alds[waveM * 16 + hx][kk];
        }
        v16bf_t af = __builtin_bit_cast(v16bf_t, a);

#pragma unroll
        for (int ns = 0; ns < 2; ++ns) {      // B fragments: 32x16 each
            v8u_t bv;
#pragma unroll
            for (int v = 0; v < 8; ++v)
                bv[v] = *(const unsigned*)&Blds[waveN * 32 + ns * 16 + hx][16 * g + 2 * v];
            v16bf_t bfr = __builtin_bit_cast(v16bf_t, bv);
            if (ns == 0)
                acc0 = __builtin_amdgcn_wmma_f32_16x16x32_bf16(false, af, false, bfr,
                                                               (short)0, acc0, false, false);
            else
                acc1 = __builtin_amdgcn_wmma_f32_16x16x32_bf16(false, af, false, bfr,
                                                               (short)0, acc1, false, false);
        }
    }

    // Epilogue: C/D layout -> row = 8*g + r, col = lane&15
#pragma unroll
    for (int ns = 0; ns < 2; ++ns) {
        v8f_t acc = ns ? acc1 : acc0;
        int col = n0 + waveN * 32 + ns * 16 + hx;
        float bvl = bias[col];
#pragma unroll
        for (int r = 0; r < 8; ++r) {
            int row = m0 + waveM * 16 + 8 * g + r;
            out[(size_t)row * Nd + col] = f2bf(acc[r] + bvl);
        }
    }
}

// ---------------------------------------------------------------------------
// Fused attention: one block = 16 query rows of one batch.
// Q tile + double-buffered K tiles staged by the Tensor Data Mover
// (tensor_load_to_lds + s_wait_tensorcnt), overlapping DMA with WMMA+logf.
// scores = Q K^T * scale + log(cw+1e-8); softmax; write attn; colsum atomics.
// ---------------------------------------------------------------------------
__global__ __launch_bounds__(256) void attn_kernel(
    const unsigned short* __restrict__ Qb, const unsigned short* __restrict__ Kb,
    const float* __restrict__ cw, float* __restrict__ attn_out,
    float* __restrict__ colsum)
{
    __shared__ unsigned short Qlds[16][256];     // 8 KB
    __shared__ unsigned short Klds[2][128][32];  // 2 x 8 KB double buffer
    __shared__ float S[16][1024];                // 64 KB (WGP has 320 KB)

    const int tid  = threadIdx.x;
    const int lane = tid & 31, wave = tid >> 5;
    const int b  = blockIdx.x >> 6;
    const int n0 = (blockIdx.x & 63) * 16;
    const int g = lane >> 4, hx = lane & 15;

    const unsigned short* krow0 = Kb + (size_t)(b * Ndim) * HDim;
    const unsigned qoff  = lds_offset_of(&Qlds[0][0]);
    const unsigned koff0 = lds_offset_of(&Klds[0][0][0]);
    const unsigned koff1 = lds_offset_of(&Klds[1][0][0]);

    if (wave == 0) {
        // Q tile: 16 rows x 256 cols (contiguous rows)
        tdm_load_2d(qoff, Qb + (size_t)(b * Ndim + n0) * HDim,
                    256, 16, 256, Bdim * Ndim, 256);
        // first K tile: 128 rows x 32 cols (t = 0 -> mt=0, kc=0)
        tdm_load_2d(koff0, krow0, 32, 128, 256, Bdim * Ndim, 256);
    }

    const float scale = 0.0625f;                 // 1/sqrt(256)
    v8f_t acc = {};

    for (int t = 0; t < 64; ++t) {               // t = mt*8 + kc
        const int mt = t >> 3, kc = t & 7;
        if (kc == 0) acc = (v8f_t){};

        if (wave == 0) {
            if (t < 63) {                        // prefetch tile t+1 into other buffer
                int t1 = t + 1;
                tdm_load_2d((t1 & 1) ? koff1 : koff0,
                            krow0 + (size_t)((t1 >> 3) * 128) * HDim + (t1 & 7) * 32,
                            32, 128, 256, Bdim * Ndim, 256);
                __builtin_amdgcn_s_wait_tensorcnt(1);   // tile t done, t+1 in flight
            } else {
                __builtin_amdgcn_s_wait_tensorcnt(0);
            }
        }
        __syncthreads();                         // release consumers of tile t

        v8u_t a;
#pragma unroll
        for (int v = 0; v < 8; ++v) {
            int kk = 16 * g + 2 * v;
            a[v] = *(const unsigned*)&Qlds[hx][kc * 32 + kk];
        }
        v16bf_t af = __builtin_bit_cast(v16bf_t, a);
        v8u_t bv;
#pragma unroll
        for (int v = 0; v < 8; ++v)
            bv[v] = *(const unsigned*)&Klds[t & 1][wave * 16 + hx][16 * g + 2 * v];
        v16bf_t bfr = __builtin_bit_cast(v16bf_t, bv);
        acc = __builtin_amdgcn_wmma_f32_16x16x32_bf16(false, af, false, bfr,
                                                      (short)0, acc, false, false);

        if (kc == 7) {                           // scores chunk complete
            int col = mt * 128 + wave * 16 + hx;
#pragma unroll
            for (int r = 0; r < 8; ++r) {
                int row = 8 * g + r;
                float c = cw[((size_t)(b * Ndim) + n0 + row) * Ndim + col];
                S[row][col] = acc[r] * scale + __logf(c + 1e-8f);
            }
        }
        __syncthreads();                         // tile t fully consumed before reuse
    }

    {   // row softmax: 16 lanes per row, shfl_xor reductions in 16-lane groups
        int row = tid >> 4, l = tid & 15;
        float mx = -3.4e38f;
        for (int j = 0; j < 64; ++j) mx = fmaxf(mx, S[row][l + 16 * j]);
#pragma unroll
        for (int m = 8; m >= 1; m >>= 1) mx = fmaxf(mx, __shfl_xor(mx, m, 16));
        float sum = 0.f;
        for (int j = 0; j < 64; ++j) {
            float e = __expf(S[row][l + 16 * j] - mx);
            S[row][l + 16 * j] = e;
            sum += e;
        }
#pragma unroll
        for (int m = 8; m >= 1; m >>= 1) sum += __shfl_xor(sum, m, 16);
        float inv = 1.0f / sum;
        float* orow = attn_out + ((size_t)(b * Ndim) + n0 + row) * Ndim;
        for (int j = 0; j < 64; ++j) {
            float v = S[row][l + 16 * j] * inv;
            S[row][l + 16 * j] = v;
            orow[l + 16 * j] = v;                // attn to d_out (coalesced per 16-group)
        }
    }
    __syncthreads();

    // column partial sums for pooling:  mean_n(attn@V) = (colsum/N) @ V
#pragma unroll
    for (int i = 0; i < 4; ++i) {
        int c = tid + 256 * i;
        float s = 0.f;
#pragma unroll
        for (int r = 0; r < 16; ++r) s += S[r][c];
        atomicAdd(&colsum[b * Ndim + c], s);
    }
}

// ---------------------------------------------------------------------------
// Small fp32 kernels
// ---------------------------------------------------------------------------
__global__ void zero_kernel(float* p, int n) {
    int i = blockIdx.x * 256 + threadIdx.x;
    if (i < n) p[i] = 0.f;
}

__global__ void cpool_kernel(const float* __restrict__ cf, float* __restrict__ cpool) {
    int i = blockIdx.x * 256 + threadIdx.x;           // 32*896
    if (i >= Bdim * CDim) return;
    int b = i / CDim, c = i % CDim;
    const float* p = cf + (size_t)b * Ndim * CDim + c;
    float s = 0.f;
    for (int n = 0; n < Ndim; ++n) s += p[(size_t)n * CDim];
    cpool[i] = s * (1.0f / Ndim);
}

__global__ void gpool_kernel(const float* __restrict__ colsum,
                             const unsigned short* __restrict__ Vb,
                             float* __restrict__ gpool) {
    int i = blockIdx.x * 256 + threadIdx.x;           // 32*256
    if (i >= Bdim * HDim) return;
    int b = i >> 8, d = i & 255;
    const unsigned short* vp = Vb + (size_t)b * Ndim * HDim + d;
    const float* cs = colsum + b * Ndim;
    float s = 0.f;
    for (int m = 0; m < Ndim; ++m) s += cs[m] * bf2f(vp[(size_t)m * HDim]);
    gpool[i] = s * (1.0f / Ndim);
}

__global__ void combined_kernel(const float* __restrict__ cpool, const float* __restrict__ gpool,
                                const float* __restrict__ Wcp, const float* __restrict__ bcp,
                                const float* __restrict__ Wgp, const float* __restrict__ bgp,
                                float* __restrict__ comb) {
    int i = blockIdx.x * 256 + threadIdx.x;           // 32*1280
    if (i >= Bdim * ODim) return;
    int b = i / ODim, c = i % ODim;
    float s;
    if (c < ODim / 2) {
        s = bcp[c];
        const float* x = cpool + b * CDim;
        for (int k = 0; k < CDim; ++k) s += x[k] * Wcp[k * (ODim / 2) + c];
    } else {
        int c2 = c - ODim / 2;
        s = bgp[c2];
        const float* x = gpool + b * HDim;
        for (int k = 0; k < HDim; ++k) s += x[k] * Wgp[k * (ODim / 2) + c2];
    }
    comb[i] = s;
}

__global__ void head1_kernel(const float* __restrict__ comb, const float* __restrict__ Wf1,
                             const float* __restrict__ bf1, const float* __restrict__ gmm,
                             const float* __restrict__ bet, const float* __restrict__ mean,
                             const float* __restrict__ var, float* __restrict__ h) {
    int i = blockIdx.x * 256 + threadIdx.x;           // 32*1280
    if (i >= Bdim * ODim) return;
    int b = i / ODim, c = i % ODim;
    float s = bf1[c];
    const float* x = comb + b * ODim;
    for (int k = 0; k < ODim; ++k) s += x[k] * Wf1[k * ODim + c];
    s = (s - mean[c]) * rsqrtf(var[c] + 1e-5f) * gmm[c] + bet[c];
    h[i] = fmaxf(s, 0.f);
}

__global__ void head2_kernel(const float* __restrict__ h, const float* __restrict__ Wf2,
                             const float* __restrict__ bf2w, float* __restrict__ out) {
    int i = blockIdx.x * 256 + threadIdx.x;           // 32*1280
    if (i >= Bdim * ODim) return;
    int b = i / ODim, c = i % ODim;
    float s = bf2w[c];
    const float* x = h + b * ODim;
    for (int k = 0; k < ODim; ++k) s += x[k] * Wf2[k * ODim + c];
    out[i] = s;
}

// ---------------------------------------------------------------------------
extern "C" void kernel_launch(void* const* d_in, const int* in_sizes, int n_in,
                              void* d_out, int out_size, void* d_ws, size_t ws_size,
                              hipStream_t stream) {
    const float* cf  = (const float*)d_in[0];
    const float* gf  = (const float*)d_in[1];
    const float* cw  = (const float*)d_in[2];
    const float* Wq  = (const float*)d_in[3];  const float* bq  = (const float*)d_in[4];
    const float* Wk  = (const float*)d_in[5];  const float* bk  = (const float*)d_in[6];
    const float* Wv  = (const float*)d_in[7];  const float* bv  = (const float*)d_in[8];
    const float* Wcp = (const float*)d_in[9];  const float* bcp = (const float*)d_in[10];
    const float* Wgp = (const float*)d_in[11]; const float* bgp = (const float*)d_in[12];
    const float* Wf1 = (const float*)d_in[13]; const float* bf1 = (const float*)d_in[14];
    const float* gmm = (const float*)d_in[15]; const float* bet = (const float*)d_in[16];
    const float* mean= (const float*)d_in[17]; const float* var = (const float*)d_in[18];
    const float* Wf2 = (const float*)d_in[19]; const float* bf2w= (const float*)d_in[20];

    const size_t M = (size_t)Bdim * Ndim;             // 32768 rows
    char* ws = (char*)d_ws;
    size_t off = 0;
    unsigned short* Qb = (unsigned short*)(ws + off); off += M * HDim * 2;
    unsigned short* Kb = (unsigned short*)(ws + off); off += M * HDim * 2;
    unsigned short* Vb = (unsigned short*)(ws + off); off += M * HDim * 2;
    float* colsum = (float*)(ws + off); off += (size_t)Bdim * Ndim * 4;
    float* cpool  = (float*)(ws + off); off += (size_t)Bdim * CDim * 4;
    float* gpool  = (float*)(ws + off); off += (size_t)Bdim * HDim * 4;
    float* comb   = (float*)(ws + off); off += (size_t)Bdim * ODim * 4;
    float* hbuf   = (float*)(ws + off); off += (size_t)Bdim * ODim * 4;

    float* out_head = (float*)d_out;
    float* attn_out = out_head + (size_t)Bdim * ODim; // tuple: (out, attn)

    zero_kernel<<<(Bdim * Ndim + 255) / 256, 256, 0, stream>>>(colsum, Bdim * Ndim);

    dim3 gproj((unsigned)(M / 64), HDim / 64);        // (512, 4)
    gemm_bf16_kernel<<<gproj, 256, 0, stream>>>(cf, Wq, bq, Qb, (int)M, CDim, HDim);
    gemm_bf16_kernel<<<gproj, 256, 0, stream>>>(gf, Wk, bk, Kb, (int)M, GDim, HDim);
    gemm_bf16_kernel<<<gproj, 256, 0, stream>>>(gf, Wv, bv, Vb, (int)M, GDim, HDim);

    cpool_kernel<<<(Bdim * CDim + 255) / 256, 256, 0, stream>>>(cf, cpool);

    attn_kernel<<<Bdim * (Ndim / 16), 256, 0, stream>>>(Qb, Kb, cw, attn_out, colsum);

    gpool_kernel<<<(Bdim * HDim + 255) / 256, 256, 0, stream>>>(colsum, Vb, gpool);
    combined_kernel<<<(Bdim * ODim + 255) / 256, 256, 0, stream>>>(cpool, gpool, Wcp, bcp,
                                                                   Wgp, bgp, comb);
    head1_kernel<<<(Bdim * ODim + 255) / 256, 256, 0, stream>>>(comb, Wf1, bf1, gmm, bet,
                                                                mean, var, hbuf);
    head2_kernel<<<(Bdim * ODim + 255) / 256, 256, 0, stream>>>(hbuf, Wf2, bf2w, out_head);
}
